// CausalSelfAttention_28544352649362
// MI455X (gfx1250) — matrix operand order repaired and checked by reading, not compiled, not run on hardware
//
#include <hip/hip_runtime.h>
#include <math.h>
#include <stdint.h>

// ---------------------------------------------------------------------------
// CausalSelfAttention on MI455X (gfx1250, wave32, WMMA)
//   x[2,2048,1024] f32, Wqkv[3072,1024] f32, Wout[1024,1024] f32 -> out f32
// bf16 WMMA (v_wmma_f32_16x16x32_bf16) everywhere, fp32 accumulate.
// GEMMs: async double-buffered LDS staging (GLOBAL_LOAD_ASYNC_TO_LDS_B128 +
// s_wait_asynccnt). Attention: flash tiling, scores never leave LDS/VGPRs.
// ---------------------------------------------------------------------------

#define Bq   2
#define Tq   2048
#define Cq   1024
#define Hq   16
#define Dh   64
#define Mrows (Bq * Tq)      // 4096
#define ThreeC (3 * Cq)      // 3072

typedef __attribute__((ext_vector_type(16))) __bf16 v16bf;
typedef __attribute__((ext_vector_type(8)))  float  v8f;

union Frag { v16bf v; uint4 u[2]; };

// pointee type the async builtin expects: int __vector(4)
typedef int v4i_t __attribute__((vector_size(16)));
typedef __attribute__((address_space(1))) v4i_t gv4i;   // global
typedef __attribute__((address_space(3))) v4i_t lv4i;   // LDS

static __device__ __forceinline__ unsigned short f2bf(float x) {
    unsigned u = __float_as_uint(x);
    u += 0x7fffu + ((u >> 16) & 1u);          // round-to-nearest-even
    return (unsigned short)(u >> 16);
}

static __device__ __forceinline__ v8f wmma_bf16(v16bf a, v16bf b, v8f c) {
    return __builtin_amdgcn_wmma_f32_16x16x32_bf16(
        false, a, false, b, (short)0, c, false, false);
}

// ---- CDNA5 async global->LDS (ASYNCcnt-tracked), 16 bytes per lane --------
static __device__ __forceinline__ void async_g2l_b128(const void* g, void* l) {
#if __has_builtin(__builtin_amdgcn_global_load_async_to_lds_b128)
    __builtin_amdgcn_global_load_async_to_lds_b128(
        (gv4i*)g, (lv4i*)l, 0, 0);
#else
    unsigned laddr = (unsigned)(uintptr_t)l;               // DS byte address
    unsigned long long gaddr = (unsigned long long)(uintptr_t)g;
    asm volatile("global_load_async_to_lds_b128 %0, %1, off"
                 :: "v"(laddr), "v"(gaddr) : "memory");
#endif
}

static __device__ __forceinline__ void wait_async0() {
#if __has_builtin(__builtin_amdgcn_s_wait_asynccnt)
    __builtin_amdgcn_s_wait_asynccnt(0);
#else
    asm volatile("s_wait_asynccnt 0x0" ::: "memory");
#endif
}

// A fragment (16x32 bf16): lane m=L%16; VGPR0-3 = K 8*(L/16)..+7, VGPR4-7 = +16
static __device__ __forceinline__ Frag load_fragA(const unsigned short* row_base,
                                                  int k0, int lhalf) {
    Frag f;
    const unsigned short* p = row_base + k0 + lhalf * 8;
    f.u[0] = *(const uint4*)(p);
    f.u[1] = *(const uint4*)(p + 16);
    return f;
}

// B fragment (32x16 bf16) from B^T rows: lane n=L%16; VGPR0-3 = K 8*(L/16)..+7, VGPR4-7 = +16
static __device__ __forceinline__ Frag load_fragB(const unsigned short* nrow_base,
                                                  int k0, int lhalf) {
    Frag f;
    const unsigned short* p = nrow_base + k0 + lhalf * 8;
    f.u[0] = *(const uint4*)(p);
    f.u[1] = *(const uint4*)(p + 16);
    return f;
}

// ---------------------------------------------------------------------------
// fp32 -> bf16 conversion
// ---------------------------------------------------------------------------
__global__ void k_cvt_bf16(const float* __restrict__ s,
                           unsigned short* __restrict__ d, int n) {
    int i = blockIdx.x * blockDim.x + threadIdx.x;
    int stride = gridDim.x * blockDim.x;
    for (; i < n; i += stride) d[i] = f2bf(s[i]);
}

// ---------------------------------------------------------------------------
// GEMM: C[m,n] = sum_k A[m,k] * W[n,k]   (A: MxK bf16 rm, W: NxK bf16 rm)
// 256 thr = 8 waves (2M x 4N), block tile 128x128, each wave 64x32, K-step 32.
// A/B 128x32 tiles staged in LDS by async DMA, double-buffered:
//   issue async for buf^1  ->  WMMA from buf  ->  s_wait_asynccnt 0 -> barrier
// LDS rows padded to 40 elems (80B) to spread DS banks.
// ---------------------------------------------------------------------------
#define TSTR 40   // padded LDS row stride (elements) for 32-element rows

template <bool OUT_BF16>
__global__ void k_gemm(const unsigned short* __restrict__ A,
                       const unsigned short* __restrict__ W,
                       void* __restrict__ Cout, int M, int N, int K) {
    __shared__ __align__(16) unsigned short Abuf[2][128 * TSTR];
    __shared__ __align__(16) unsigned short Bbuf[2][128 * TSTR];

    const int tid   = threadIdx.x;
    const int lane  = tid & 31;
    const int wid   = tid >> 5;
    const int lmod  = lane & 15;
    const int lhalf = lane >> 4;
    const int wm    = wid >> 2;                 // 0..1
    const int wn    = wid & 3;                  // 0..3
    const int m0 = blockIdx.y * 128;
    const int n0 = blockIdx.x * 128;

    // tile copy: 8KB = 512 x 16B chunks; thread owns chunks tid and tid+256
    // chunk c -> row c>>2, element offset (c&3)*8
    const int ar0 = tid >> 2,         ao0 = (tid & 3) * 8;
    const int ar1 = (tid + 256) >> 2, ao1 = ((tid + 256) & 3) * 8;

    v8f acc[4][2];
    for (int i = 0; i < 4; ++i)
        for (int j = 0; j < 2; ++j) acc[i][j] = (v8f){0,0,0,0,0,0,0,0};

    const int nK = K / 32;
    int cur = 0;

    // prime buffer 0
    {
        async_g2l_b128(A + (size_t)(m0 + ar0) * K + ao0, &Abuf[0][ar0 * TSTR + ao0]);
        async_g2l_b128(A + (size_t)(m0 + ar1) * K + ao1, &Abuf[0][ar1 * TSTR + ao1]);
        async_g2l_b128(W + (size_t)(n0 + ar0) * K + ao0, &Bbuf[0][ar0 * TSTR + ao0]);
        async_g2l_b128(W + (size_t)(n0 + ar1) * K + ao1, &Bbuf[0][ar1 * TSTR + ao1]);
    }
    wait_async0();
    __syncthreads();

    for (int kt = 0; kt < nK; ++kt) {
        const int nxt = cur ^ 1;
        if (kt + 1 < nK) {
            const int k0 = (kt + 1) * 32;
            async_g2l_b128(A + (size_t)(m0 + ar0) * K + k0 + ao0, &Abuf[nxt][ar0 * TSTR + ao0]);
            async_g2l_b128(A + (size_t)(m0 + ar1) * K + k0 + ao1, &Abuf[nxt][ar1 * TSTR + ao1]);
            async_g2l_b128(W + (size_t)(n0 + ar0) * K + k0 + ao0, &Bbuf[nxt][ar0 * TSTR + ao0]);
            async_g2l_b128(W + (size_t)(n0 + ar1) * K + k0 + ao1, &Bbuf[nxt][ar1 * TSTR + ao1]);
        }

        // compute from cur (ds_load_b128 fragment reads)
        Frag a[4], b[2];
        for (int i = 0; i < 4; ++i) {
            const unsigned short* p = &Abuf[cur][(wm * 64 + i * 16 + lmod) * TSTR];
            a[i] = load_fragA(p, 0, lhalf);
        }
        for (int j = 0; j < 2; ++j) {
            const unsigned short* p = &Bbuf[cur][(wn * 32 + j * 16 + lmod) * TSTR];
            b[j] = load_fragB(p, 0, lhalf);
        }
        for (int i = 0; i < 4; ++i)
            for (int j = 0; j < 2; ++j)
                acc[i][j] = wmma_bf16(a[i].v, b[j].v, acc[i][j]);

        wait_async0();       // our prefetch for nxt landed (overlapped with WMMA)
        __syncthreads();     // all waves done prefetching nxt / reading cur
        cur = nxt;
    }

    // C/D layout: N = lane%16, M = r + 8*(lane/16)
    for (int i = 0; i < 4; ++i)
        for (int j = 0; j < 2; ++j)
            for (int r = 0; r < 8; ++r) {
                int row = m0 + wm * 64 + i * 16 + r + 8 * lhalf;
                int col = n0 + wn * 32 + j * 16 + lmod;
                if (OUT_BF16)
                    ((unsigned short*)Cout)[(size_t)row * N + col] = f2bf(acc[i][j][r]);
                else
                    ((float*)Cout)[(size_t)row * N + col] = acc[i][j][r];
            }
}

// ---------------------------------------------------------------------------
// Repack QKV[4096,3072] bf16 -> per-head Q[bh][t][d], K[bh][t][d], V^T[bh][d][t]
// ---------------------------------------------------------------------------
__global__ void k_repack(const unsigned short* __restrict__ qkv,
                         unsigned short* __restrict__ Qh,
                         unsigned short* __restrict__ Kh,
                         unsigned short* __restrict__ Vt) {
    long long i = (long long)blockIdx.x * blockDim.x + threadIdx.x;
    long long n = (long long)Mrows * ThreeC;
    long long stride = (long long)gridDim.x * blockDim.x;
    for (; i < n; i += stride) {
        int m = (int)(i / ThreeC), f = (int)(i % ThreeC);
        int b = m / Tq, t = m % Tq;
        int which = f / Cq, g = f % Cq;
        int h = g / Dh, d = g % Dh;
        unsigned short v = qkv[i];
        int bh = b * Hq + h;
        if (which == 0)      Qh[((size_t)bh * Tq + t) * Dh + d] = v;
        else if (which == 1) Kh[((size_t)bh * Tq + t) * Dh + d] = v;
        else                 Vt[((size_t)bh * Dh + d) * Tq + t] = v;
    }
}

// ---------------------------------------------------------------------------
// Flash attention. Grid = B*H*(T/64) blocks, 128 threads (4 waves).
// Each wave: 16 query rows, full Dh=64 fp32 accumulator (4 x v8f).
// S = QK^T via WMMA -> causal mask + 1/8 scale -> LDS -> online softmax
// (LDS round trip also converts C-layout to A-layout) -> O = O*alpha + P V.
// ---------------------------------------------------------------------------
__global__ void k_flash(const unsigned short* __restrict__ Qh,
                        const unsigned short* __restrict__ Kh,
                        const unsigned short* __restrict__ Vt,
                        unsigned short* __restrict__ Att) {
    const int lane  = threadIdx.x & 31;
    const int w     = threadIdx.x >> 5;
    const int lmod  = lane & 15;
    const int lhalf = lane >> 4;

    const int blk = blockIdx.x;
    const int qb  = blk & 31;
    const int bh  = blk >> 5;
    const int b   = bh >> 4;
    const int h   = bh & 15;
    const int qrow0 = qb * 64 + w * 16;

    const size_t baseQK = (size_t)bh * Tq * Dh;
    const size_t baseV  = (size_t)bh * Dh * Tq;

    __shared__ __align__(16) float          sbuf[4][16 * 68];
    __shared__ __align__(16) unsigned short pbuf[4][16 * 72];
    __shared__ float m_s[4][16], l_s[4][16], sc_s[4][16];

    Frag aq[2];
    {
        const unsigned short* p = Qh + baseQK + (size_t)(qrow0 + lmod) * Dh;
        aq[0] = load_fragA(p, 0, lhalf);
        aq[1] = load_fragA(p, 32, lhalf);
    }

    if (lane < 16) { m_s[w][lane] = -__builtin_inff(); l_s[w][lane] = 0.0f; }

    v8f acc[4];
    for (int dt = 0; dt < 4; ++dt) acc[dt] = (v8f){0,0,0,0,0,0,0,0};
    __syncthreads();

    for (int kb = 0; kb <= qb; ++kb) {
        // prefetch next block's K/V rows into cache (global_prefetch_b8)
        if (kb < qb) {
            __builtin_prefetch(Kh + baseQK + (size_t)((kb + 1) * 64 + lane * 2) * Dh, 0, 3);
            __builtin_prefetch(Vt + baseV + (size_t)(lane * 2) * Tq + (kb + 1) * 64, 0, 3);
        }

        // ---- S = (Q K^T) * 1/sqrt(Dh), causal-masked, -> LDS ----
        for (int nt = 0; nt < 4; ++nt) {
            const unsigned short* pk =
                Kh + baseQK + (size_t)(kb * 64 + nt * 16 + lmod) * Dh;
            Frag bk0 = load_fragB(pk, 0, lhalf);
            Frag bk1 = load_fragB(pk, 32, lhalf);
            v8f s = (v8f){0,0,0,0,0,0,0,0};
            s = wmma_bf16(aq[0].v, bk0.v, s);
            s = wmma_bf16(aq[1].v, bk1.v, s);
            const int col  = nt * 16 + lmod;
            const int kpos = kb * 64 + col;
            for (int r = 0; r < 8; ++r) {
                const int q = qrow0 + r + 8 * lhalf;
                float v = s[r] * 0.125f;
                if (kpos > q) v = -__builtin_inff();
                sbuf[w][(r + 8 * lhalf) * 68 + col] = v;
            }
        }
        __syncthreads();

        // ---- online softmax: lanes 0..15 own one query row each ----
        if (lane < 16) {
            float mo = m_s[w][lane], lo = l_s[w][lane];
            const float* srow = &sbuf[w][lane * 68];
            float bmax = -__builtin_inff();
            for (int j = 0; j < 64; ++j) bmax = fmaxf(bmax, srow[j]);
            float mn    = fmaxf(mo, bmax);
            float alpha = __expf(mo - mn);
            float ls = 0.0f;
            unsigned short* prow = &pbuf[w][lane * 72];
            for (int j = 0; j < 64; ++j) {
                float p = __expf(srow[j] - mn);
                ls += p;
                prow[j] = f2bf(p);
            }
            m_s[w][lane]  = mn;
            l_s[w][lane]  = lo * alpha + ls;
            sc_s[w][lane] = alpha;
        }
        __syncthreads();

        // ---- O = O*alpha + P V ----
        float sc[8];
        for (int r = 0; r < 8; ++r) sc[r] = sc_s[w][8 * lhalf + r];

        Frag ap[2];
        {
            const unsigned short* pp = &pbuf[w][lmod * 72];
            ap[0] = load_fragA(pp, 0, lhalf);
            ap[1] = load_fragA(pp, 32, lhalf);
        }
        for (int dt = 0; dt < 4; ++dt) {
            for (int r = 0; r < 8; ++r) acc[dt][r] *= sc[r];
            const unsigned short* pv =
                Vt + baseV + (size_t)(dt * 16 + lmod) * Tq + kb * 64;
            Frag bv0 = load_fragB(pv, 0, lhalf);
            Frag bv1 = load_fragB(pv, 32, lhalf);
            acc[dt] = wmma_bf16(ap[0].v, bv0.v, acc[dt]);
            acc[dt] = wmma_bf16(ap[1].v, bv1.v, acc[dt]);
        }
        __syncthreads();
    }

    float li[8];
    for (int r = 0; r < 8; ++r) li[r] = 1.0f / l_s[w][8 * lhalf + r];
    for (int dt = 0; dt < 4; ++dt)
        for (int r = 0; r < 8; ++r) {
            int trow = qrow0 + r + 8 * lhalf;
            int col  = h * Dh + dt * 16 + lmod;
            Att[(size_t)(b * Tq + trow) * Cq + col] = f2bf(acc[dt][r] * li[r]);
        }
}

// ---------------------------------------------------------------------------
// Launch. Workspace (64 MB):
//   [0,8M) Xb (reused as Att), [8M,14M) Wqkvb, [14M,16M) Woutb,
//   [16M,40M) QKVb, [40M,48M) Qh, [48M,56M) Kh, [56M,64M) Vt
// ---------------------------------------------------------------------------
extern "C" void kernel_launch(void* const* d_in, const int* in_sizes, int n_in,
                              void* d_out, int out_size, void* d_ws, size_t ws_size,
                              hipStream_t stream) {
    (void)in_sizes; (void)n_in; (void)out_size; (void)ws_size;
    const float* x    = (const float*)d_in[0];
    const float* Wqkv = (const float*)d_in[1];
    const float* Wout = (const float*)d_in[2];
    float* out = (float*)d_out;

    char* ws = (char*)d_ws;
    unsigned short* Xb    = (unsigned short*)(ws);
    unsigned short* Wqkvb = (unsigned short*)(ws + (8ull << 20));
    unsigned short* Woutb = (unsigned short*)(ws + (14ull << 20));
    unsigned short* QKVb  = (unsigned short*)(ws + (16ull << 20));
    unsigned short* Qh    = (unsigned short*)(ws + (40ull << 20));
    unsigned short* Kh    = (unsigned short*)(ws + (48ull << 20));
    unsigned short* Vt    = (unsigned short*)(ws + (56ull << 20));
    unsigned short* Att   = Xb;   // Xb dead after GEMM1

    k_cvt_bf16<<<2048, 256, 0, stream>>>(x, Xb, Mrows * Cq);
    k_cvt_bf16<<<2048, 256, 0, stream>>>(Wqkv, Wqkvb, ThreeC * Cq);
    k_cvt_bf16<<<2048, 256, 0, stream>>>(Wout, Woutb, Cq * Cq);

    k_gemm<true><<<dim3(ThreeC / 128, Mrows / 128), 256, 0, stream>>>(
        Xb, Wqkvb, QKVb, Mrows, ThreeC, Cq);

    k_repack<<<4096, 256, 0, stream>>>(QKVb, Qh, Kh, Vt);

    k_flash<<<Bq * Hq * (Tq / 64), 128, 0, stream>>>(Qh, Kh, Vt, Att);

    k_gemm<false><<<dim3(Cq / 128, Mrows / 128), 256, 0, stream>>>(
        Att, Woutb, out, Mrows, Cq, Cq);
}
